// BERTHeading_14551349199352
// MI455X (gfx1250) — matrix operand order repaired and checked by reading, not compiled, not run on hardware
//
#include <hip/hip_runtime.h>

typedef __bf16 bf16_t;
typedef bf16_t v4bf  __attribute__((ext_vector_type(4)));
typedef bf16_t v8bf  __attribute__((ext_vector_type(8)));
typedef bf16_t v16bf __attribute__((ext_vector_type(16)));
typedef float  v8f   __attribute__((ext_vector_type(8)));

#define EMB  768
#define FEAT 512
#define SEQ  20
#define KT_N 24          // 768 / 32 k-steps
#define NT_N 32          // 512 / 16 n-tiles per slice
#define SLICE_ELEMS ((size_t)NT_N * KT_N * 512)   // packed bf16 elems per slice

// ---------------------------------------------------------------------------
// Pack weights into wave32 WMMA B-fragment layout.
// Fragment (slice, nt, kt): lane holds col n = nt*16 + (lane&15),
// K = kt*32 + (lane>>4)*8 + j (j<8), kt*32 + 16 + (lane>>4)*8 + (j-8) (j>=8), stored contiguously (32B/lane).
// Slices: 0=W1[0], 1=W2[0], 2=W2[1], 3=W3[0], 4=W3[1], 5=W3[2], 6=Wp.
// ---------------------------------------------------------------------------
__global__ void pack_weights(const float* __restrict__ W1,
                             const float* __restrict__ W2,
                             const float* __restrict__ W3,
                             const float* __restrict__ Wp,
                             bf16_t* __restrict__ pk) {
  int gid = blockIdx.x * blockDim.x + threadIdx.x;     // 7*32*24*32 lanes
  if (gid >= 7 * NT_N * KT_N * 32) return;
  int lane  = gid & 31;
  int frag  = gid >> 5;                                 // (slice*32+nt)*24+kt
  int kt    = frag % KT_N;
  int nt    = (frag / KT_N) % NT_N;
  int slice = frag / (KT_N * NT_N);
  const float* src;
  switch (slice) {
    case 0: src = W1;                      break;
    case 1: src = W2;                      break;
    case 2: src = W2 + (size_t)EMB*FEAT;   break;
    case 3: src = W3;                      break;
    case 4: src = W3 + (size_t)EMB*FEAT;   break;
    case 5: src = W3 + (size_t)2*EMB*FEAT; break;
    default: src = Wp;                     break;
  }
  int kbase = kt * 32 + (lane >> 4) * 8;
  int n     = nt * 16 + (lane & 15);
  bf16_t* dst = pk + ((size_t)frag * 512 + (size_t)lane * 16);
#pragma unroll
  for (int j = 0; j < 16; ++j)
    dst[j] = (bf16_t)src[(size_t)(kbase + j + (j >= 8 ? 8 : 0)) * FEAT + n];
}

// ---------------------------------------------------------------------------
// Main fused kernel: per workgroup = 4 samples (80 rows, 5 M-tiles), 6 waves.
// 16 passes over channel groups; each pass computes P[2 groups][80][6][16],
// combines (shift-add + bias + relu + max), writes unnormalized outputs and
// accumulates sum-of-squares / pooled maxima.
// ---------------------------------------------------------------------------
#define XS_STRIDE 776    // 768 + 8 bf16 pad: 16B-aligned rows, bank-spread

__launch_bounds__(192, 1)
__global__ void bert_main(const float* __restrict__ x,      // words_emb
                          const float* __restrict__ b1,
                          const float* __restrict__ b2,
                          const float* __restrict__ b3,
                          const bf16_t* __restrict__ pk,
                          float* __restrict__ out_words,     // [bs][512][20]
                          float* __restrict__ out_wv,        // [bs][512]
                          float* __restrict__ ws_ssw,        // [bs*20]
                          float* __restrict__ ws_ssv) {      // [bs]
  __shared__ bf16_t Xs[80 * XS_STRIDE];        // 124,160 B
  __shared__ bf16_t Pb[2 * 80 * 6 * 16];       //  30,720 B
  __shared__ float  ssc[80];
  __shared__ float  ssv[4];
  __shared__ int    pmax[3][4][32];

  const int tid   = threadIdx.x;
  const int lane  = tid & 31;
  const int w     = tid >> 5;           // wave 0..5
  const int bbase = blockIdx.x * 4;     // first sample of this WG

  // init reductions
  for (int i = tid; i < 80; i += 192) ssc[i] = 0.f;
  if (tid < 4) ssv[tid] = 0.f;
  for (int i = tid; i < 3 * 4 * 32; i += 192) (&pmax[0][0][0])[i] = 0;

  // stage X tile as bf16 into LDS (each element converted exactly once)
  const float* xg = x + (size_t)bbase * SEQ * EMB;
  for (int i = tid; i < 80 * EMB / 4; i += 192) {
    float4 f = ((const float4*)xg)[i];
    int r = i / (EMB / 4);
    int k = (i % (EMB / 4)) * 4;
    v4bf v;
    v[0] = (bf16_t)f.x; v[1] = (bf16_t)f.y; v[2] = (bf16_t)f.z; v[3] = (bf16_t)f.w;
    *(v4bf*)&Xs[r * XS_STRIDE + k] = v;
  }
  __syncthreads();

  const int gi   = w / 3;               // channel group within pass (0/1)
  const int s0   = (w % 3) * 2;         // this wave's slice pair
  const int lrow = lane & 15;
  const int lk   = (lane >> 4) * 8;     // A k-offset for this lane half

  for (int pass = 0; pass < 16; ++pass) {
    const int nt = pass * 2 + gi;       // channel n-tile index 0..31
    const bf16_t* pb0 = pk + (size_t)((s0 * NT_N + nt) * KT_N) * 512 + (size_t)lane * 16;
    const bf16_t* pb1 = pb0 + SLICE_ELEMS;

    v8f acc[5][2];
#pragma unroll
    for (int m = 0; m < 5; ++m) { acc[m][0] = (v8f)0.f; acc[m][1] = (v8f)0.f; }

    for (int kt = 0; kt < KT_N; ++kt) {
      v16bf B0 = *(const v16bf*)(pb0 + (size_t)kt * 512);
      v16bf B1 = *(const v16bf*)(pb1 + (size_t)kt * 512);
      // load all 5 A fragments into distinct registers first so the DS loads
      // issue as one clause and WMMAs only take partial dscnt waits
      union { v16bf v; v8bf h[2]; } au[5];
#pragma unroll
      for (int m = 0; m < 5; ++m) {
        int xi = (m * 16 + lrow) * XS_STRIDE + kt * 32 + lk;
        au[m].h[0] = *(const v8bf*)&Xs[xi];
        au[m].h[1] = *(const v8bf*)&Xs[xi + 16];
      }
#pragma unroll
      for (int m = 0; m < 5; ++m) {
        acc[m][0] = __builtin_amdgcn_wmma_f32_16x16x32_bf16(
            false, au[m].v, false, B0, (short)0, acc[m][0], false, false);
        acc[m][1] = __builtin_amdgcn_wmma_f32_16x16x32_bf16(
            false, au[m].v, false, B1, (short)0, acc[m][1], false, false);
      }
    }

    // scatter accumulators (C layout: VGPR r -> M = r + 8*(lane>>4), N = lane&15)
    {
      const int mhi = (lane >> 4) * 8;
      const int n   = lane & 15;
#pragma unroll
      for (int m = 0; m < 5; ++m)
#pragma unroll
        for (int r = 0; r < 8; ++r) {
          int row = m * 16 + r + mhi;
          Pb[((gi * 80 + row) * 6 + s0)     * 16 + n] = (bf16_t)acc[m][0][r];
          Pb[((gi * 80 + row) * 6 + s0 + 1) * 16 + n] = (bf16_t)acc[m][1][r];
        }
    }
    __syncthreads();   // P ready (also fences previous pass's consumers)

    // combine: shift-add + bias + relu + positional max + reductions
    for (int idx = tid; idx < 160; idx += 192) {
      int g   = idx / 80;
      int row = idx - g * 80;
      int smp = row / SEQ;
      int l   = row - smp * SEQ;
      const bf16_t* Pr = &Pb[(g * 80 + row) * 96];
      float  ssum  = 0.f;
      size_t obase = (((size_t)(bbase + smp)) * FEAT + (size_t)(pass * 2 + g) * 16) * SEQ + l;
#pragma unroll
      for (int ch = 0; ch < 16; ++ch) {
        int   cg = (pass * 2 + g) * 16 + ch;
        float va = fmaxf((float)Pr[ch] + b1[cg], 0.f);
        float vb = (l < 19)
          ? fmaxf((float)Pr[16 + ch] + (float)Pr[96 + 32 + ch] + b2[cg], 0.f) : 0.f;
        float vc = (l < 18)
          ? fmaxf((float)Pr[48 + ch] + (float)Pr[96 + 64 + ch] +
                  (float)Pr[192 + 80 + ch] + b3[cg], 0.f) : 0.f;
        float code = fmaxf(va, fmaxf(vb, vc));
        out_words[obase + (size_t)ch * SEQ] = code;     // unnormalized
        ssum += code * code;
        int cc = g * 16 + ch;
        atomicMax(&pmax[0][smp][cc], __float_as_int(va));  // relu >= 0
        atomicMax(&pmax[1][smp][cc], __float_as_int(vb));
        atomicMax(&pmax[2][smp][cc], __float_as_int(vc));
      }
      atomicAdd(&ssc[row], ssum);
    }
    __syncthreads();   // pmax complete

    // pooled word feature for the 32 channels of this pass
    if (tid < 128) {
      int smp = tid >> 5, cc = tid & 31;
      float ma = __int_as_float(pmax[0][smp][cc]);
      float mb = __int_as_float(pmax[1][smp][cc]);
      float mc = __int_as_float(pmax[2][smp][cc]);
      float pooled = (ma + mb + mc) * (1.f / 3.f);
      out_wv[(size_t)(bbase + smp) * FEAT + pass * 32 + cc] = pooled;  // unnormalized
      atomicAdd(&ssv[smp], pooled * pooled);
      pmax[0][smp][cc] = 0; pmax[1][smp][cc] = 0; pmax[2][smp][cc] = 0;
    }
  }

  __syncthreads();
  if (tid < 80) ws_ssw[(size_t)blockIdx.x * 80 + tid] = ssc[tid];
  if (tid < 4)  ws_ssv[bbase + tid] = ssv[tid];
}

// ---------------------------------------------------------------------------
// Sentence projection: [bs,768] x [768,512] + bp via WMMA (packed slice 6).
// ---------------------------------------------------------------------------
__launch_bounds__(128, 1)
__global__ void sent_proj(const float* __restrict__ s_emb,
                          const float* __restrict__ bp,
                          const bf16_t* __restrict__ pk,
                          float* __restrict__ out_sent) {
  const int lane = threadIdx.x & 31;
  const int w    = threadIdx.x >> 5;
  const int mt   = blockIdx.x;                 // m-tile (16 rows)
  const int nt   = blockIdx.y * 4 + w;         // n-tile 0..31
  const int row  = mt * 16 + (lane & 15);
  const int lk   = (lane >> 4) * 8;
  const bf16_t* pb = pk + (size_t)((6 * NT_N + nt) * KT_N) * 512 + (size_t)lane * 16;

  v8f acc = (v8f)0.f;
  for (int kt = 0; kt < KT_N; ++kt) {
    const float* xr = s_emb + (size_t)row * EMB + kt * 32 + lk;
    float4 f0 = *(const float4*)(xr);
    float4 f1 = *(const float4*)(xr + 4);
    float4 f2 = *(const float4*)(xr + 16);
    float4 f3 = *(const float4*)(xr + 20);
    union { v16bf v; v8bf h[2]; } au;
    au.v[0]  = (bf16_t)f0.x; au.v[1]  = (bf16_t)f0.y;
    au.v[2]  = (bf16_t)f0.z; au.v[3]  = (bf16_t)f0.w;
    au.v[4]  = (bf16_t)f1.x; au.v[5]  = (bf16_t)f1.y;
    au.v[6]  = (bf16_t)f1.z; au.v[7]  = (bf16_t)f1.w;
    au.v[8]  = (bf16_t)f2.x; au.v[9]  = (bf16_t)f2.y;
    au.v[10] = (bf16_t)f2.z; au.v[11] = (bf16_t)f2.w;
    au.v[12] = (bf16_t)f3.x; au.v[13] = (bf16_t)f3.y;
    au.v[14] = (bf16_t)f3.z; au.v[15] = (bf16_t)f3.w;
    v16bf B = *(const v16bf*)(pb + (size_t)kt * 512);
    acc = __builtin_amdgcn_wmma_f32_16x16x32_bf16(
        false, au.v, false, B, (short)0, acc, false, false);
  }
  const int n = nt * 16 + (lane & 15);
  const float bias = bp[n];
#pragma unroll
  for (int r = 0; r < 8; ++r) {
    int ro = mt * 16 + r + (lane >> 4) * 8;
    out_sent[(size_t)ro * FEAT + n] = acc[r] + bias;
  }
}

// ---------------------------------------------------------------------------
// Final L2 normalization of words_out and word_vector (x / max(||x||, 1e-12)).
// ---------------------------------------------------------------------------
__global__ void norm_kernel(float* __restrict__ out,
                            const float* __restrict__ ssw,
                            const float* __restrict__ ssv,
                            size_t N1, size_t N2) {
  size_t idx = (size_t)blockIdx.x * blockDim.x + threadIdx.x;
  if (idx < N1) {
    size_t b = idx / ((size_t)FEAT * SEQ);
    int    l = (int)(idx % SEQ);
    float ss = ssw[b * SEQ + l];
    out[idx] *= 1.f / fmaxf(sqrtf(ss), 1e-12f);
  } else if (idx < N1 + N2) {
    size_t j = idx - N1;
    float ss = ssv[j / FEAT];
    out[idx] *= 1.f / fmaxf(sqrtf(ss), 1e-12f);
  }
}

// ---------------------------------------------------------------------------
extern "C" void kernel_launch(void* const* d_in, const int* in_sizes, int n_in,
                              void* d_out, int out_size, void* d_ws, size_t ws_size,
                              hipStream_t stream) {
  const float* words = (const float*)d_in[0];
  const float* sent  = (const float*)d_in[1];
  const float* W1    = (const float*)d_in[2];
  const float* b1    = (const float*)d_in[3];
  const float* W2    = (const float*)d_in[4];
  const float* b2    = (const float*)d_in[5];
  const float* W3    = (const float*)d_in[6];
  const float* b3    = (const float*)d_in[7];
  const float* Wp    = (const float*)d_in[8];
  const float* bp    = (const float*)d_in[9];

  const int bs = in_sizes[0] / (SEQ * EMB);          // 4096
  float* out = (float*)d_out;
  const size_t N1 = (size_t)bs * FEAT * SEQ;         // words_out
  const size_t N2 = (size_t)bs * FEAT;               // word_vector

  bf16_t* pk = (bf16_t*)d_ws;
  const size_t PKB = (size_t)7 * SLICE_ELEMS * sizeof(bf16_t);   // ~5.5 MB
  float* ssw = (float*)((char*)d_ws + PKB);
  float* ssv = ssw + (size_t)bs * SEQ;

  // 1) pack all 7 weight slices into WMMA B-fragment layout
  {
    int total = 7 * NT_N * KT_N * 32;
    pack_weights<<<(total + 255) / 256, 256, 0, stream>>>(W1, W2, W3, Wp, pk);
  }
  // 2) fused conv-as-GEMM + combine
  bert_main<<<dim3(bs / 4), dim3(192), 0, stream>>>(
      words, b1, b2, b3, pk, out, out + N1, ssw, ssv);
  // 3) sentence projection
  sent_proj<<<dim3(bs / 16, 8), dim3(128), 0, stream>>>(
      sent, bp, pk, out + N1 + N2);
  // 4) L2 normalization
  {
    size_t total = N1 + N2;
    norm_kernel<<<(unsigned)((total + 255) / 256), 256, 0, stream>>>(
        out, ssw, ssv, N1, N2);
  }
}
